// mLSTMLayer_8744553415140
// MI455X (gfx1250) — matrix operand order repaired and checked by reading, not compiled, not run on hardware
//
#include <hip/hip_runtime.h>
#include <hip/hip_bf16.h>

// ---------------------------------------------------------------------------
// mLSTM layer (xLSTM) for MI455X / gfx1250.
// bf16 WMMA GEMMs with LDS double-buffering via async global->LDS loads.
// B=2, S=1024, E=2048, NH=8, DH=256, KVH=1
// ---------------------------------------------------------------------------

constexpr int cB  = 2;
constexpr int cS  = 1024;
constexpr int cE  = 2048;
constexpr int cNH = 8;
constexpr int cDH = 256;          // E / NH
constexpr int cM  = cB * cS;      // 2048 tokens
constexpr float cCAP = 15.0f;
constexpr float cEPS = 1e-6f;
#define NEG_BIG (-1e30f)

// Toggle for the CDNA5 async global->LDS staging path.
#define USE_ASYNC 1

typedef __attribute__((ext_vector_type(16))) __bf16 v16bf;
typedef __attribute__((ext_vector_type(8)))  __bf16 v8bf;
typedef __attribute__((ext_vector_type(8)))  float  v8f;

__device__ __forceinline__ __bf16 f2bf(float f) {
    unsigned u = __builtin_bit_cast(unsigned, f);
    unsigned r = u + 0x7FFFu + ((u >> 16) & 1u);  // RNE
    unsigned short h = (unsigned short)(r >> 16);
    return __builtin_bit_cast(__bf16, h);
}

// One 16-byte chunk: global -> LDS.
// Async path: GLOBAL_LOAD_ASYNC_TO_LDS_B128 (ASYNCcnt-tracked, ISA 15.18.3 op 98).
// VDST = LDS byte offset (low 32 bits of generic shared pointer), VADDR = global addr.
__device__ __forceinline__ void stage_chunk(const __bf16* g, __bf16* l) {
#if USE_ASYNC
    unsigned lds_off = (unsigned)(unsigned long long)l;
    asm volatile("global_load_async_to_lds_b128 %0, %1, off"
                 :: "v"(lds_off), "v"(g) : "memory");
#else
    *(uint4*)l = *(const uint4*)g;
#endif
}

__device__ __forceinline__ void wait_async_le4() {
#if USE_ASYNC
    asm volatile("s_wait_asynccnt 0x4" ::: "memory");
#endif
}
__device__ __forceinline__ void wait_async_le0() {
#if USE_ASYNC
    asm volatile("s_wait_asynccnt 0x0" ::: "memory");
#endif
}

// ---------------------------------------------------------------------------
// Elementwise f32 -> bf16
// ---------------------------------------------------------------------------
__global__ void cvt_f32_bf16(const float* __restrict__ src, __bf16* __restrict__ dst, int n) {
    int i = blockIdx.x * blockDim.x + threadIdx.x;
    if (i < n) dst[i] = f2bf(src[i]);
}

// ---------------------------------------------------------------------------
// Transpose + convert: src f32 [R, C] row-major -> dst bf16 [C, R] row-major
// ---------------------------------------------------------------------------
__global__ void tr_f32_bf16(const float* __restrict__ src, __bf16* __restrict__ dst, int R, int C) {
    int i = blockIdx.x * blockDim.x + threadIdx.x;
    if (i >= R * C) return;
    int r = i / C, c = i % C;
    dst[(size_t)c * R + r] = f2bf(src[(size_t)r * C + c]);
}

// ---------------------------------------------------------------------------
// Tiled GEMM: C[M,N] (f32) = A[M,K](bf16) * B, BT is [N,K] bf16.
// 256 threads = 8 waves; workgroup tile 128x128; wave tile 32x64 (2x4 WMMA tiles).
// K-chunks of 32 staged into double-buffered LDS with async global->LDS loads.
// Fragment layouts used by this code (same contiguous k order in A and B):
//   A-frag (16x32): lane l = M row, elems e: K = 16h + e
//   B-frag (32x16): lane l = N col, elems e: K = 16h + e
//   C tile: VGPR r, row = 8h + r, col = l
// ---------------------------------------------------------------------------
__global__ void __launch_bounds__(256)
gemm_bf16_wmma_tiled(const __bf16* __restrict__ A, const __bf16* __restrict__ BT,
                     float* __restrict__ C, int M, int N, int K) {
    __shared__ alignas(32) __bf16 ldsA[2][128 * 32];
    __shared__ alignas(32) __bf16 ldsB[2][128 * 32];

    const int tid  = threadIdx.x;
    const int lane = tid & 31;
    const int wave = tid >> 5;
    const int half = lane >> 4;
    const int l    = lane & 15;
    const int wm   = wave & 3;            // 0..3 : 32 M-rows each
    const int wn   = wave >> 2;           // 0..1 : 64 N-cols each
    const int tm0  = blockIdx.y * 128;
    const int tn0  = blockIdx.x * 128;
    const int KC   = K >> 5;

    v8f acc[2][4];
#pragma unroll
    for (int mi = 0; mi < 2; ++mi)
#pragma unroll
        for (int ni = 0; ni < 4; ++ni) acc[mi][ni] = v8f{};

    // Stage one K-chunk (A block 128x32, B block 128x32) into LDS buffer `buf`.
    // 512 x 16B chunks per matrix; each thread stages 2 chunks per matrix
    // => 4 async ops per thread per stage.
    auto stage = [&](int buf, int k0) {
#pragma unroll
        for (int ii = 0; ii < 2; ++ii) {
            int chunk = tid + ii * 256;
            int rr = chunk >> 2;          // 0..127
            int c8 = (chunk & 3) * 8;     // 0,8,16,24
            stage_chunk(A  + (size_t)(tm0 + rr) * K + k0 + c8, &ldsA[buf][rr * 32 + c8]);
            stage_chunk(BT + (size_t)(tn0 + rr) * K + k0 + c8, &ldsB[buf][rr * 32 + c8]);
        }
    };

    stage(0, 0);
    for (int kc = 0; kc < KC; ++kc) {
        const int buf = kc & 1;
        if (kc + 1 < KC) {
            stage(buf ^ 1, (kc + 1) * 32);
            wait_async_le4();             // retire current buffer, next may stay in flight
        } else {
            wait_async_le0();
        }
        __syncthreads();

        v16bf afr[2], bfr[4];
#pragma unroll
        for (int mi = 0; mi < 2; ++mi) {
            int row = wm * 32 + mi * 16 + l;
            afr[mi] = *(const v16bf*)&ldsA[buf][row * 32 + 16 * half];
        }
#pragma unroll
        for (int ni = 0; ni < 4; ++ni) {
            int col = wn * 64 + ni * 16 + l;
            bfr[ni] = *(const v16bf*)&ldsB[buf][col * 32 + 16 * half];
        }
#pragma unroll
        for (int mi = 0; mi < 2; ++mi)
#pragma unroll
            for (int ni = 0; ni < 4; ++ni)
                acc[mi][ni] = __builtin_amdgcn_wmma_f32_16x16x32_bf16(
                                  false, afr[mi], false, bfr[ni], (short)0,
                                  acc[mi][ni], false, false);
        __syncthreads();
    }

#pragma unroll
    for (int mi = 0; mi < 2; ++mi)
#pragma unroll
        for (int ni = 0; ni < 4; ++ni) {
            int col = tn0 + wn * 64 + ni * 16 + l;
            float* crow = C + (size_t)(tm0 + wm * 32 + mi * 16 + 8 * half) * N + col;
#pragma unroll
            for (int r = 0; r < 8; ++r) crow[(size_t)r * N] = acc[mi][ni][r];
        }
}

// ---------------------------------------------------------------------------
// Gate projections i/f: per (token, head) dot over E, + bias, soft cap.
// ip/fp layout [B, NH, S]
// ---------------------------------------------------------------------------
__global__ void gates_kernel(const float* __restrict__ x,
                             const float* __restrict__ Wi, const float* __restrict__ bi,
                             const float* __restrict__ Wf, const float* __restrict__ bfg,
                             float* __restrict__ ip, float* __restrict__ fp) {
    int idx = blockIdx.x * blockDim.x + threadIdx.x;     // B*S*NH
    if (idx >= cM * cNH) return;
    int h = idx % cNH;
    int tok = idx / cNH;
    const float* xr = x + (size_t)tok * cE;
    float si = bi[h], sf = bfg[h];
    for (int e = 0; e < cE; ++e) {
        float xv = xr[e];
        si += xv * Wi[(size_t)e * cNH + h];
        sf += xv * Wf[(size_t)e * cNH + h];
    }
    si = cCAP * tanhf(si / cCAP);
    sf = cCAP * tanhf(sf / cCAP);
    int b = tok / cS, s = tok % cS;
    ip[((size_t)b * cNH + h) * cS + s] = si;
    fp[((size_t)b * cNH + h) * cS + s] = sf;
}

// ---------------------------------------------------------------------------
// Serial cumsum of log-sigmoid(f_pre) per (b,h). 16 tiny scans.
// ---------------------------------------------------------------------------
__global__ void scan_kernel(const float* __restrict__ fp, float* __restrict__ cf) {
    if (threadIdx.x != 0) return;
    int bh = blockIdx.x;                      // 0 .. B*NH-1
    const float* f = fp + (size_t)bh * cS;
    float* c = cf + (size_t)bh * cS;
    float acc = 0.0f;
    for (int s = 0; s < cS; ++s) {
        float z = f[s];
        float nz = -z;
        float sp = (nz > 20.0f) ? nz : log1pf(expf(nz));  // softplus(-z)
        acc += -sp;                                        // log_sigmoid(z)
        c[s] = acc;
    }
}

// ---------------------------------------------------------------------------
// Flash-style mLSTM attention. One wave per (b, head, 16-row query tile).
//  qb : bf16 [B, S, NH, DH]   kb : bf16 [B, S, DH]   vT : bf16 [B, DH, S]
//  cf/ip : f32 [B, NH, S]     hout : f32 [B, S, NH, DH]
// ---------------------------------------------------------------------------
__global__ void __launch_bounds__(32)
mlstm_attn_kernel(const __bf16* __restrict__ qb, const __bf16* __restrict__ kb,
                  const __bf16* __restrict__ vT, const float* __restrict__ cf,
                  const float* __restrict__ ip, float* __restrict__ hout) {
    __shared__ alignas(32) __bf16 plds[16 * 32];

    const int lane = threadIdx.x & 31;
    const int half = lane >> 4;
    const int l    = lane & 15;
    const int qt = blockIdx.x;     // S/16
    const int hd = blockIdx.y;     // NH
    const int b  = blockIdx.z;     // B
    const int s0 = qt * 16;

    const float* cfr = cf + ((size_t)b * cNH + hd) * cS;
    const float* ipr = ip + ((size_t)b * cNH + hd) * cS;

    // Cache q A-fragments for the whole DH (8 K-chunks of 32).
    v16bf qfrag[8];
    {
        const __bf16* qrow = qb + ((size_t)((b * cS + s0 + l) * cNH + hd)) * cDH;
#pragma unroll
        for (int kc = 0; kc < 8; ++kc) {
            qfrag[kc] = *(const v16bf*)(qrow + kc * 32 + 16 * half);
        }
    }

    float m_run[8], n_run[8], cfq[8];
#pragma unroll
    for (int r = 0; r < 8; ++r) {
        m_run[r] = NEG_BIG;
        n_run[r] = 0.0f;
        cfq[r]   = cfr[s0 + 8 * half + r];
    }
    v8f hacc[16];
#pragma unroll
    for (int c = 0; c < 16; ++c) hacc[c] = v8f{};

    const float inv_sqrt_dh = 0.0625f;   // 1/sqrt(256)
    const __bf16* vbase = vT + (size_t)b * cDH * cS;
    const __bf16* kbase = kb + (size_t)b * cS * cDH;

    for (int t0 = 0; t0 <= s0 + 15; t0 += 32) {
        v8f ct[2];
#pragma unroll
        for (int j = 0; j < 2; ++j) {
            v8f c = {};
            const __bf16* krow = kbase + (size_t)(t0 + j * 16 + l) * cDH;
#pragma unroll
            for (int kc = 0; kc < 8; ++kc) {
                v16bf bfrag = *(const v16bf*)(krow + kc * 32 + 16 * half);
                c = __builtin_amdgcn_wmma_f32_16x16x32_bf16(
                        false, qfrag[kc], false, bfrag, (short)0, c, false, false);
            }
            ct[j] = c;
        }

        float cfk[2], ipk[2];
#pragma unroll
        for (int j = 0; j < 2; ++j) {
            int t = t0 + j * 16 + l;
            cfk[j] = cfr[t];
            ipk[j] = ipr[t];
        }
        float mblk[8];
        float ld[2][8];
#pragma unroll
        for (int r = 0; r < 8; ++r) mblk[r] = NEG_BIG;
#pragma unroll
        for (int j = 0; j < 2; ++j) {
            int t = t0 + j * 16 + l;
#pragma unroll
            for (int r = 0; r < 8; ++r) {
                int srow = s0 + 8 * half + r;
                float v = (t <= srow) ? (cfq[r] - cfk[j] + ipk[j]) : NEG_BIG;
                ld[j][r] = v;
                mblk[r] = fmaxf(mblk[r], v);
            }
        }
#pragma unroll
        for (int off = 1; off < 16; off <<= 1)
#pragma unroll
            for (int r = 0; r < 8; ++r)
                mblk[r] = fmaxf(mblk[r], __shfl_xor(mblk[r], off, 32));

        float alpha[8], psum[8];
#pragma unroll
        for (int r = 0; r < 8; ++r) {
            float mn = fmaxf(m_run[r], mblk[r]);
            alpha[r] = expf(m_run[r] - mn);
            m_run[r] = mn;
            psum[r] = 0.0f;
        }
        __bf16 pb[2][8];
#pragma unroll
        for (int j = 0; j < 2; ++j)
#pragma unroll
            for (int r = 0; r < 8; ++r) {
                float p = ct[j][r] * inv_sqrt_dh * expf(ld[j][r] - m_run[r]);
                psum[r] += p;
                pb[j][r] = f2bf(p);
            }
#pragma unroll
        for (int off = 1; off < 16; off <<= 1)
#pragma unroll
            for (int r = 0; r < 8; ++r)
                psum[r] += __shfl_xor(psum[r], off, 32);
#pragma unroll
        for (int r = 0; r < 8; ++r)
            n_run[r] = n_run[r] * alpha[r] + psum[r];
#pragma unroll
        for (int c = 0; c < 16; ++c)
#pragma unroll
            for (int r = 0; r < 8; ++r)
                hacc[c][r] *= alpha[r];

        // re-layout p (C-layout f32 -> A-layout bf16) via LDS
#pragma unroll
        for (int j = 0; j < 2; ++j)
#pragma unroll
            for (int r = 0; r < 8; ++r)
                plds[(8 * half + r) * 32 + j * 16 + l] = pb[j][r];
        __syncthreads();
        v16bf pfrag = *(const v16bf*)(&plds[l * 32 + 16 * half]);
        __syncthreads();

#pragma unroll
        for (int c = 0; c < 16; ++c) {
            v16bf vfrag = *(const v16bf*)(vbase + (size_t)(c * 16 + l) * cS + t0 + 16 * half);
            hacc[c] = __builtin_amdgcn_wmma_f32_16x16x32_bf16(
                          false, pfrag, false, vfrag, (short)0, hacc[c], false, false);
        }
    }

#pragma unroll
    for (int r = 0; r < 8; ++r) {
        float vn = fmaxf(fabsf(n_run[r]), expf(-m_run[r]));
        float inv = 1.0f / (vn + cEPS);
        int s = s0 + 8 * half + r;
        float* orow = hout + ((size_t)((b * cS + s) * cNH + hd)) * cDH;
#pragma unroll
        for (int c = 0; c < 16; ++c)
            orow[c * 16 + l] = hacc[c][r] * inv;
    }
}

// ---------------------------------------------------------------------------
// Per-(token,head) layernorm over DH + gamma + sigmoid(o_preact) gate -> bf16.
// ---------------------------------------------------------------------------
__global__ void __launch_bounds__(256)
ln_gate_kernel(const float* __restrict__ hin,   // [B,S,NH,DH]
               const float* __restrict__ og,    // [B,S,E]
               const float* __restrict__ gamma, // [E] viewed [NH,DH]
               __bf16* __restrict__ gb) {       // [B,S,E]
    int tok = blockIdx.x;
    int hd = threadIdx.x >> 5;
    int lane = threadIdx.x & 31;
    const float* hrow = hin + ((size_t)tok * cNH + hd) * cDH;

    float vals[8];
    float sum = 0.0f, sum2 = 0.0f;
#pragma unroll
    for (int i = 0; i < 8; ++i) {
        float v = hrow[lane + 32 * i];
        vals[i] = v;
        sum += v;
        sum2 += v * v;
    }
#pragma unroll
    for (int off = 1; off < 32; off <<= 1) {
        sum  += __shfl_xor(sum,  off, 32);
        sum2 += __shfl_xor(sum2, off, 32);
    }
    float mu  = sum * (1.0f / cDH);
    float var = sum2 * (1.0f / cDH) - mu * mu;
    float rstd = rsqrtf(var + cEPS);

    const float* ogrow = og + (size_t)tok * cE + hd * cDH;
    const float* gr = gamma + hd * cDH;
    __bf16* out = gb + (size_t)tok * cE + hd * cDH;
#pragma unroll
    for (int i = 0; i < 8; ++i) {
        int c = lane + 32 * i;
        float hn = (vals[i] - mu) * rstd * gr[c];
        float o = ogrow[c];
        float sig = 1.0f / (1.0f + expf(-o));
        out[c] = f2bf(sig * hn);
    }
}

// ---------------------------------------------------------------------------
// Host launch
// ---------------------------------------------------------------------------
extern "C" void kernel_launch(void* const* d_in, const int* in_sizes, int n_in,
                              void* d_out, int out_size, void* d_ws, size_t ws_size,
                              hipStream_t stream) {
    (void)in_sizes; (void)n_in; (void)out_size; (void)ws_size;
    const float* x     = (const float*)d_in[0];
    const float* Wq    = (const float*)d_in[1];
    const float* Wk    = (const float*)d_in[2];
    const float* Wv    = (const float*)d_in[3];
    const float* Wog   = (const float*)d_in[4];
    const float* Wi    = (const float*)d_in[5];
    const float* bi    = (const float*)d_in[6];
    const float* Wf    = (const float*)d_in[7];
    const float* bfg   = (const float*)d_in[8];
    const float* gamma = (const float*)d_in[9];
    const float* Wout  = (const float*)d_in[10];
    float* y = (float*)d_out;

    char* p = (char*)d_ws;
    auto alloc = [&](size_t bytes) -> void* {
        void* r = (void*)p;
        p += (bytes + 255) & ~(size_t)255;
        return r;
    };
    __bf16* xb    = (__bf16*)alloc((size_t)cM * cE * 2);
    __bf16* WqT   = (__bf16*)alloc((size_t)cE * cE * 2);
    __bf16* WkT   = (__bf16*)alloc((size_t)cE * cDH * 2);
    __bf16* WvT   = (__bf16*)alloc((size_t)cE * cDH * 2);
    __bf16* WogT  = (__bf16*)alloc((size_t)cE * cE * 2);
    __bf16* WoutT = (__bf16*)alloc((size_t)cE * cE * 2);
    float*  qf    = (float*)alloc((size_t)cM * cE * 4);
    float*  kf    = (float*)alloc((size_t)cM * cDH * 4);
    float*  vf    = (float*)alloc((size_t)cM * cDH * 4);
    float*  ogf   = (float*)alloc((size_t)cM * cE * 4);
    __bf16* qbuf  = (__bf16*)alloc((size_t)cM * cE * 2);
    __bf16* kbuf  = (__bf16*)alloc((size_t)cM * cDH * 2);
    __bf16* vTb   = (__bf16*)alloc((size_t)cB * cDH * cS * 2);
    float*  ipb   = (float*)alloc((size_t)cB * cNH * cS * 4);
    float*  fpb   = (float*)alloc((size_t)cB * cNH * cS * 4);
    float*  cfb   = (float*)alloc((size_t)cB * cNH * cS * 4);
    float*  hbuf  = (float*)alloc((size_t)cM * cE * 4);
    __bf16* gbuf  = (__bf16*)alloc((size_t)cM * cE * 2);

    const int TB = 256;
    auto blocks = [](long n, int tb) { return (unsigned)((n + tb - 1) / tb); };

    // 1) inputs -> bf16
    cvt_f32_bf16<<<blocks((long)cM * cE, TB), TB, 0, stream>>>(x, xb, cM * cE);
    tr_f32_bf16<<<blocks((long)cE * cE, TB), TB, 0, stream>>>(Wq,   WqT,   cE, cE);
    tr_f32_bf16<<<blocks((long)cE * cDH, TB), TB, 0, stream>>>(Wk,  WkT,   cE, cDH);
    tr_f32_bf16<<<blocks((long)cE * cDH, TB), TB, 0, stream>>>(Wv,  WvT,   cE, cDH);
    tr_f32_bf16<<<blocks((long)cE * cE, TB), TB, 0, stream>>>(Wog,  WogT,  cE, cE);
    tr_f32_bf16<<<blocks((long)cE * cE, TB), TB, 0, stream>>>(Wout, WoutT, cE, cE);

    // 2) projections (tiled WMMA GEMMs, async LDS staging)
    gemm_bf16_wmma_tiled<<<dim3(cE / 128,  cM / 128), 256, 0, stream>>>(xb, WqT,  qf,  cM, cE,  cE);
    gemm_bf16_wmma_tiled<<<dim3(cDH / 128, cM / 128), 256, 0, stream>>>(xb, WkT,  kf,  cM, cDH, cE);
    gemm_bf16_wmma_tiled<<<dim3(cDH / 128, cM / 128), 256, 0, stream>>>(xb, WvT,  vf,  cM, cDH, cE);
    gemm_bf16_wmma_tiled<<<dim3(cE / 128,  cM / 128), 256, 0, stream>>>(xb, WogT, ogf, cM, cE,  cE);

    // 3) attention operand prep
    cvt_f32_bf16<<<blocks((long)cM * cE, TB), TB, 0, stream>>>(qf, qbuf, cM * cE);
    cvt_f32_bf16<<<blocks((long)cM * cDH, TB), TB, 0, stream>>>(kf, kbuf, cM * cDH);
    for (int b = 0; b < cB; ++b)
        tr_f32_bf16<<<blocks((long)cS * cDH, TB), TB, 0, stream>>>(
            vf + (size_t)b * cS * cDH, vTb + (size_t)b * cDH * cS, cS, cDH);

    // 4) gates + scan
    gates_kernel<<<blocks((long)cM * cNH, TB), TB, 0, stream>>>(x, Wi, bi, Wf, bfg, ipb, fpb);
    scan_kernel<<<cB * cNH, 32, 0, stream>>>(fpb, cfb);

    // 5) mLSTM attention (WMMA flash loop)
    mlstm_attn_kernel<<<dim3(cS / 16, cNH, cB), 32, 0, stream>>>(
        qbuf, kbuf, vTb, cfb, ipb, hbuf);

    // 6) per-head layernorm + output gate
    ln_gate_kernel<<<cM, 256, 0, stream>>>(hbuf, ogf, gamma, gbuf);

    // 7) output projection
    gemm_bf16_wmma_tiled<<<dim3(cE / 128, cM / 128), 256, 0, stream>>>(gbuf, WoutT, y, cM, cE, cE);
}
